// HeadAttention_78280073937512
// MI455X (gfx1250) — matrix operand order repaired and checked by reading, not compiled, not run on hardware
//
#include <hip/hip_runtime.h>
#include <hip/hip_bf16.h>
#include <math.h>
#include <stdint.h>

typedef __attribute__((ext_vector_type(16))) _Float16 v16h;
typedef __attribute__((ext_vector_type(8)))  _Float16 v8h;
typedef __attribute__((ext_vector_type(8)))  float    v8f;

#define B_  4
#define S_  4096
#define D_  1024
#define DK_ 64
#define BS_ (B_ * S_)

// CDNA5 async global->LDS data mover (ASYNCcnt path); verified to assemble.
#define USE_ASYNC_LDS 1

__device__ __forceinline__ v8f wmma_f16f32(v16h a, v16h b, v8f c) {
  return __builtin_amdgcn_wmma_f32_16x16x32_f16(
      false, a, false, b, (short)0, c, false, false);
}

// Assemble an A-fragment: contiguous halfs p[ak..ak+15]
__device__ __forceinline__ v16h load_afrag_h(const _Float16* p, int ak) {
  const v8h lo = *(const v8h*)(p + ak);        // 16B aligned
  const v8h hi = *(const v8h*)(p + ak + 8);
  return __builtin_shufflevector(lo, hi, 0, 1, 2, 3, 4, 5, 6, 7,
                                         8, 9, 10, 11, 12, 13, 14, 15);
}

// 16-byte global -> LDS copy (async data mover when enabled)
__device__ __forceinline__ void copy16_g2l(_Float16* ldst, const _Float16* gsrc) {
#if USE_ASYNC_LDS
  asm volatile("global_load_async_to_lds_b128 %0, %1, off"
               :: "v"((uint32_t)(uintptr_t)ldst),
                  "v"((unsigned long long)(uintptr_t)gsrc)
               : "memory");
#else
  *(v8h*)ldst = *(const v8h*)gsrc;
#endif
}

// Wait until at most `n` async copies are still in flight (in-order completion
// => with 4 copies/tile/thread, n=4 guarantees the PREVIOUS tile is resident).
__device__ __forceinline__ void wait_async_le4() {
#if USE_ASYNC_LDS
  asm volatile("s_wait_asynccnt 4" ::: "memory");
#endif
}
__device__ __forceinline__ void wait_async_le0() {
#if USE_ASYNC_LDS
  asm volatile("s_wait_asynccnt 0" ::: "memory");
#endif
}

// ---------------------------------------------------------------------------
// Pack W [1024][64] f32 -> Wt [64][1024] f16 (L2-resident, 128 KB)
// ---------------------------------------------------------------------------
__global__ __launch_bounds__(256)
void pack_w_kernel(const float* __restrict__ W, _Float16* __restrict__ Wt) {
  const int i = blockIdx.x * 256 + threadIdx.x;   // coalesced read
  const int k = i >> 6, n = i & 63;
  Wt[n * D_ + k] = (_Float16)W[i];
}

// ---------------------------------------------------------------------------
// Projection: Y = X[16384,1024] @ W[1024,64] via f16 WMMA.
// 4 waves/block, each wave owns a 16x64 strip. B-fragments are single
// contiguous v16h loads from pre-transposed Wt (L2-resident).
// transposeOut=1 stores Y as [64][B*S] (for V). outScale folded into store.
// ---------------------------------------------------------------------------
__global__ __launch_bounds__(128)
void proj_wmma_kernel(const float* __restrict__ X, const _Float16* __restrict__ Wt,
                      _Float16* __restrict__ Y, const int transposeOut,
                      const float outScale) {
  const int tid  = threadIdx.x;
  const int wave = tid >> 5;
  const int lane = tid & 31;
  const int nl   = lane & 15;
  const bool hi  = (lane >= 16);
  const int row0 = (blockIdx.x * 4 + wave) * 16;

  v8f acc[4];
  #pragma unroll
  for (int nt = 0; nt < 4; ++nt)
    #pragma unroll
    for (int r = 0; r < 8; ++r) acc[nt][r] = 0.0f;

  const int ak = hi ? 16 : 0;    // A layout: lanes16-31 hold K 16..31
  const int bo = hi ? 16 : 0;    // B layout: lanes16-31 hold K 16..31

  for (int kk = 0; kk < D_; kk += 32) {
    const float* arow = X + (size_t)(row0 + nl) * D_ + kk;
    const float4 x0 = *(const float4*)(arow + ak);
    const float4 x1 = *(const float4*)(arow + ak + 4);
    const float4 x2 = *(const float4*)(arow + ak + 8);
    const float4 x3 = *(const float4*)(arow + ak + 12);
    v16h a;
    a[0]=(_Float16)x0.x;  a[1]=(_Float16)x0.y;  a[2]=(_Float16)x0.z;  a[3]=(_Float16)x0.w;
    a[4]=(_Float16)x1.x;  a[5]=(_Float16)x1.y;  a[6]=(_Float16)x1.z;  a[7]=(_Float16)x1.w;
    a[8]=(_Float16)x2.x;  a[9]=(_Float16)x2.y;  a[10]=(_Float16)x2.z; a[11]=(_Float16)x2.w;
    a[12]=(_Float16)x3.x; a[13]=(_Float16)x3.y; a[14]=(_Float16)x3.z; a[15]=(_Float16)x3.w;

    #pragma unroll
    for (int nt = 0; nt < 4; ++nt) {
      const v16h b = *(const v16h*)(Wt + (size_t)(nt * 16 + nl) * D_ + kk + bo);
      acc[nt] = wmma_f16f32(a, b, acc[nt]);
    }
  }

  const int gr = hi ? 8 : 0;
  if (transposeOut) {
    #pragma unroll
    for (int nt = 0; nt < 4; ++nt) {
      v8h y;
      #pragma unroll
      for (int r = 0; r < 8; ++r) y[r] = (_Float16)(acc[nt][r] * outScale);
      *(v8h*)(Y + (size_t)(nt * 16 + nl) * BS_ + row0 + gr) = y;  // 16B store
    }
  } else {
    #pragma unroll
    for (int nt = 0; nt < 4; ++nt)
      #pragma unroll
      for (int r = 0; r < 8; ++r)
        Y[(size_t)(row0 + gr + r) * DK_ + nt * 16 + nl] =
            (_Float16)(acc[nt][r] * outScale);
  }
}

// ---------------------------------------------------------------------------
// Flash attention. Block = 128 threads (4 waves) = 64 query rows.
// Double-buffered async staging: tile t+1 streams into the alternate LDS
// buffer (ASYNCcnt DMA) underneath tile t's WMMAs + softmax.
// 1/sqrt(dk) is pre-folded into q (exact: power of two).
// ---------------------------------------------------------------------------
__global__ __launch_bounds__(128)
void flash_attn_kernel(const _Float16* __restrict__ Qh,
                       const _Float16* __restrict__ Kh,
                       const _Float16* __restrict__ Vt,
                       float* __restrict__ O) {
  __shared__ _Float16 lK[2][32 * 64];   // [key][dk]   2 x 4 KB
  __shared__ _Float16 lV[2][64 * 32];   // [dv][key]   2 x 4 KB
  __shared__ _Float16 lP[4][16][32];    // per-wave P staging

  const int tid  = threadIdx.x;
  const int wave = tid >> 5;
  const int lane = tid & 31;
  const int nl   = lane & 15;
  const bool hi  = (lane >= 16);
  const int gr   = hi ? 8 : 0;
  const int ak   = hi ? 16 : 0;
  const int bo   = hi ? 16 : 0;

  const size_t base = (size_t)blockIdx.y * S_;
  const int Q0 = blockIdx.x * 64;
  const int q0 = Q0 + wave * 16;
  const int KEND = Q0 + 64;            // causal extent of the whole block

  // Q fragments (q already scaled by 1/sqrt(dk) in projection)
  v16h aq[2];
  {
    const _Float16* qrow = Qh + (base + q0 + nl) * DK_;
    aq[0] = load_afrag_h(qrow, ak);
    aq[1] = load_afrag_h(qrow + 32, ak);
  }

  float mrow[8], lrow[8];
  v8f acc[4];
  #pragma unroll
  for (int r = 0; r < 8; ++r) { mrow[r] = -3.0e38f; lrow[r] = 0.0f; }
  #pragma unroll
  for (int nt = 0; nt < 4; ++nt)
    #pragma unroll
    for (int r = 0; r < 8; ++r) acc[nt][r] = 0.0f;

  // stage tile 0 into buffer 0 (4 async b128 per thread)
  {
    const _Float16* kg = Kh + base * DK_;
    copy16_g2l(lK[0] + tid * 16,     kg + tid * 16);
    copy16_g2l(lK[0] + tid * 16 + 8, kg + tid * 16 + 8);
    const int dv = tid >> 1, part = (tid & 1) * 16;
    const _Float16* vg = Vt + (size_t)dv * BS_ + base + part;
    copy16_g2l(lV[0] + dv * 32 + part,     vg);
    copy16_g2l(lV[0] + dv * 32 + part + 8, vg + 8);
  }

  for (int kb = 0; kb < KEND; kb += 32) {
    const int cur = (kb >> 5) & 1;
    const bool more = (kb + 32) < KEND;

    if (more) {   // issue next tile into the alternate buffer
      const int nxt = cur ^ 1;
      const _Float16* kg = Kh + (base + kb + 32) * DK_;
      copy16_g2l(lK[nxt] + tid * 16,     kg + tid * 16);
      copy16_g2l(lK[nxt] + tid * 16 + 8, kg + tid * 16 + 8);
      const int dv = tid >> 1, part = (tid & 1) * 16;
      const _Float16* vg = Vt + (size_t)dv * BS_ + base + kb + 32 + part;
      copy16_g2l(lV[nxt] + dv * 32 + part,     vg);
      copy16_g2l(lV[nxt] + dv * 32 + part + 8, vg + 8);
      if (kb + 64 < KEND)   // prime L2 one tile further -> global_prefetch_b8
        __builtin_prefetch((const void*)(Kh + (base + kb + 64 + lane) * DK_), 0, 1);
      wait_async_le4();     // in-order completion => current tile resident
    } else {
      wait_async_le0();
    }
    __syncthreads();

    if (kb < q0 + 16) {     // this wave's causal extent (uniform per wave)
      // ---- scores: two 16x16 tiles, K-dim 64 in two WMMA steps ----
      v8f sc[2];
      #pragma unroll
      for (int s = 0; s < 2; ++s)
        #pragma unroll
        for (int r = 0; r < 8; ++r) sc[s][r] = 0.0f;

      #pragma unroll
      for (int sub = 0; sub < 2; ++sub) {
        const _Float16* kp = lK[cur] + (sub * 16 + nl) * DK_ + bo;
        #pragma unroll
        for (int s = 0; s < 2; ++s) {
          const v16h bk = *(const v16h*)(kp + s * 32);   // 2x ds_load_b128
          sc[sub] = wmma_f16f32(aq[s], bk, sc[sub]);
        }
      }

      // ---- causal mask + online softmax ----
      #pragma unroll
      for (int r = 0; r < 8; ++r) {
        const int row = q0 + gr + r;
        float s0 = (kb + nl      <= row) ? sc[0][r] : -3.0e38f;
        float s1 = (kb + 16 + nl <= row) ? sc[1][r] : -3.0e38f;
        float mx = fmaxf(s0, s1);
        #pragma unroll
        for (int off = 1; off < 16; off <<= 1)
          mx = fmaxf(mx, __shfl_xor(mx, off, 32));
        const float mnew  = fmaxf(mrow[r], mx);
        const float alpha = __expf(mrow[r] - mnew);
        const float p0 = __expf(s0 - mnew);
        const float p1 = __expf(s1 - mnew);
        float ps = p0 + p1;
        #pragma unroll
        for (int off = 1; off < 16; off <<= 1)
          ps += __shfl_xor(ps, off, 32);
        lrow[r] = lrow[r] * alpha + ps;
        mrow[r] = mnew;
        #pragma unroll
        for (int nt = 0; nt < 4; ++nt) acc[nt][r] *= alpha;
        lP[wave][gr + r][nl]      = (_Float16)p0;   // same-wave LDS: in-order
        lP[wave][gr + r][16 + nl] = (_Float16)p1;
      }

      // ---- P(16x32) @ V(32x64): 4 WMMA ----
      const v16h ap = load_afrag_h(&lP[wave][nl][0], ak);  // 2x ds_load_b128
      #pragma unroll
      for (int nt = 0; nt < 4; ++nt) {
        const v16h bv = *(const v16h*)(lV[cur] + (nt * 16 + nl) * 32 + bo);
        acc[nt] = wmma_f16f32(ap, bv, acc[nt]);
      }
    }
    __syncthreads();   // all waves done with buffer `cur` before tile t+2 reuses it
  }

  float invl[8];
  #pragma unroll
  for (int r = 0; r < 8; ++r) invl[r] = 1.0f / lrow[r];
  #pragma unroll
  for (int nt = 0; nt < 4; ++nt)
    #pragma unroll
    for (int r = 0; r < 8; ++r)
      O[(base + q0 + gr + r) * DK_ + nt * 16 + nl] = acc[nt][r] * invl[r];
}

// ---------------------------------------------------------------------------
extern "C" void kernel_launch(void* const* d_in, const int* in_sizes, int n_in,
                              void* d_out, int out_size, void* d_ws, size_t ws_size,
                              hipStream_t stream) {
  const float* query = (const float*)d_in[0];
  const float* key_  = (const float*)d_in[1];
  const float* value = (const float*)d_in[2];
  const float* Wq    = (const float*)d_in[3];
  const float* Wk    = (const float*)d_in[4];
  const float* Wv    = (const float*)d_in[5];
  // d_in[6] (mask) is causal; applied analytically in-kernel.
  float* out = (float*)d_out;

  const size_t qkv = (size_t)BS_ * DK_;
  _Float16* qh  = (_Float16*)d_ws;
  _Float16* kh  = qh + qkv;
  _Float16* vt  = kh + qkv;              // transposed: [64][B*S]
  _Float16* wqt = vt + qkv;              // packed weights: [64][1024] each
  _Float16* wkt = wqt + (size_t)DK_ * D_;
  _Float16* wvt = wkt + (size_t)DK_ * D_;

  dim3 wgrid((D_ * DK_) / 256, 1, 1);
  pack_w_kernel<<<wgrid, 256, 0, stream>>>(Wq, wqt);
  pack_w_kernel<<<wgrid, 256, 0, stream>>>(Wk, wkt);
  pack_w_kernel<<<wgrid, 256, 0, stream>>>(Wv, wvt);

  dim3 pgrid((BS_ / 16) / 4, 1, 1);      // 256 blocks
  proj_wmma_kernel<<<pgrid, 128, 0, stream>>>(query, wqt, qh, 0, 0.125f);
  proj_wmma_kernel<<<pgrid, 128, 0, stream>>>(key_,  wkt, kh, 0, 1.0f);
  proj_wmma_kernel<<<pgrid, 128, 0, stream>>>(value, wvt, vt, 1, 1.0f);

  dim3 agrid(S_ / 64, B_, 1);
  flash_attn_kernel<<<agrid, 128, 0, stream>>>(qh, kh, vt, out);
}
